// GNNModel_47115791237139
// MI455X (gfx1250) — matrix operand order repaired and checked by reading, not compiled, not run on hardware
//
#include <hip/hip_runtime.h>
#include <hip/hip_bf16.h>
#include <stdint.h>

// ---------------------------------------------------------------------------
// Types for CDNA5 WMMA (wave32): v_wmma_f32_16x16x32_bf16
// ---------------------------------------------------------------------------
typedef __bf16 bf16_t;
typedef __attribute__((ext_vector_type(16))) __bf16 v16bf;
typedef __attribute__((ext_vector_type(8)))  __bf16 v8bf;
typedef __attribute__((ext_vector_type(8)))  float  v8f;

#define HID 128

// ---------------------------------------------------------------------------
// Graph prep
// ---------------------------------------------------------------------------
__global__ void k_deg_init(float* deg, int N) {
    int i = blockIdx.x * blockDim.x + threadIdx.x;
    if (i < N) deg[i] = 1.0f;                    // self-loop contributes 1
}

__global__ void k_deg_acc(const long long* __restrict__ ei, float* deg, int E) {
    int e = blockIdx.x * blockDim.x + threadIdx.x;
    if (e < E) atomicAdd(&deg[ei[(size_t)E + e]], 1.0f);   // dst row
}

__global__ void k_deg_to_dinv(float* deg, int N) {
    int i = blockIdx.x * blockDim.x + threadIdx.x;
    if (i < N) deg[i] = rsqrtf(deg[i]);          // deg >= 1 always (self-loop)
}

// ---------------------------------------------------------------------------
// f32 -> bf16 conversions
// ---------------------------------------------------------------------------
__global__ void k_f32_to_bf16(const float* __restrict__ in, bf16_t* __restrict__ out, size_t n) {
    size_t i = (size_t)blockIdx.x * blockDim.x + threadIdx.x;
    if (i < n) out[i] = (bf16_t)in[i];
}

// W[k][n] (KxN row-major) -> WT[n][k] bf16 (transposed so B loads are contiguous in K)
__global__ void k_convW_T(const float* __restrict__ W, bf16_t* __restrict__ WT) {
    int idx = blockIdx.x * blockDim.x + threadIdx.x;      // 128*128
    if (idx >= HID * HID) return;
    int n = idx >> 7, k = idx & 127;
    WT[idx] = (bf16_t)W[k * HID + n];
}

// Pack heads [128x3 | 128x2 | 128x3] into WT16[n][k], n=0..15 (cols 8..15 zero)
__global__ void k_convHead_T(const float* __restrict__ Wa, const float* __restrict__ Ws,
                             const float* __restrict__ We, bf16_t* __restrict__ WT) {
    int idx = blockIdx.x * blockDim.x + threadIdx.x;      // 16*128
    if (idx >= 16 * HID) return;
    int n = idx >> 7, k = idx & 127;
    float v = 0.0f;
    if      (n < 3) v = Wa[k * 3 + n];
    else if (n < 5) v = Ws[k * 2 + (n - 3)];
    else if (n < 8) v = We[k * 3 + (n - 5)];
    WT[idx] = (bf16_t)v;
}

// ---------------------------------------------------------------------------
// WMMA GEMM: C[M,128] = A[M,128](bf16) * B[128,128], B given transposed bf16.
// Block = 256 threads (8 waves). Block -> 16-row tile; wave w -> 16-col tile.
// K = 128 -> 4x v_wmma_f32_16x16x32_bf16. M must be a multiple of 16 (50000 is).
// ---------------------------------------------------------------------------
__global__ void k_gemm128_wmma(const bf16_t* __restrict__ A,
                               const bf16_t* __restrict__ BT,
                               float* __restrict__ C, int M) {
    const int wave = threadIdx.x >> 5;
    const int lane = threadIdx.x & 31;
    const int h = lane >> 4;          // lane half selects K-subrange
    const int l = lane & 15;
    const int rowTile = blockIdx.x;   // 16 rows
    const int row = rowTile * 16 + l;
    const int colBase = wave * 16;
    if (row >= M) return;

    const bf16_t* Arow = A  + (size_t)row * HID;
    const bf16_t* Bcol = BT + (size_t)(colBase + l) * HID;

    v8f acc = {};
#pragma unroll
    for (int k = 0; k < 4; ++k) {
        // A 16x32: a[i]=A[row][32k+16h+i] (contiguous)
        v16bf a = *(const v16bf*)(Arow + 32 * k + 16 * h);
        // B 32x16: b[i] = B[32k+16h+i][col] = BT[col][32k+16h+i] (contiguous)
        v16bf b = *(const v16bf*)(Bcol + 32 * k + 16 * h);
        acc = __builtin_amdgcn_wmma_f32_16x16x32_bf16(false, a, false, b,
                                                      (short)0, acc, false, false);
    }
    // D f32 16x16: VGPR r -> row r+8h, col = l
    float* Crow = C + (size_t)(rowTile * 16) * HID + colBase + l;
#pragma unroll
    for (int r = 0; r < 8; ++r)
        Crow[(size_t)(r + 8 * h) * HID] = acc[r];
}

// Fused head GEMM: HH[M,8] = A[M,128](bf16) * WhT(16x128, cols 8..15 zero).
// 8 waves/block, one 16-row tile per wave, single 16x16 output tile.
__global__ void k_gemm_head_wmma(const bf16_t* __restrict__ A,
                                 const bf16_t* __restrict__ BT,
                                 float* __restrict__ HH, int M) {
    const int wave = threadIdx.x >> 5;
    const int lane = threadIdx.x & 31;
    const int h = lane >> 4;
    const int l = lane & 15;
    const int tile = blockIdx.x * 8 + wave;       // wave-uniform guard: EXEC stays full
    if (tile * 16 >= M) return;

    const bf16_t* Arow = A  + (size_t)(tile * 16 + l) * HID;
    const bf16_t* Bcol = BT + (size_t)l * HID;

    v8f acc = {};
#pragma unroll
    for (int k = 0; k < 4; ++k) {
        v16bf a = *(const v16bf*)(Arow + 32 * k + 16 * h);
        v16bf b = *(const v16bf*)(Bcol + 32 * k + 16 * h);
        acc = __builtin_amdgcn_wmma_f32_16x16x32_bf16(false, a, false, b,
                                                      (short)0, acc, false, false);
    }
    if (l < 8) {                                   // only 8 real head columns
        float* Crow = HH + (size_t)(tile * 16) * 8 + l;
#pragma unroll
        for (int r = 0; r < 8; ++r)
            Crow[(size_t)(r + 8 * h) * 8] = acc[r];
    }
}

// ---------------------------------------------------------------------------
// Aggregation: agg = selfloop + scatter-add over edges (norm = dinv[s]*dinv[d])
// ---------------------------------------------------------------------------
__global__ void k_init_self128(const float* __restrict__ hlin, const float* __restrict__ dinv,
                               float* __restrict__ agg, int N) {
    size_t idx = (size_t)blockIdx.x * blockDim.x + threadIdx.x;
    if (idx >= (size_t)N * HID) return;
    int i = (int)(idx >> 7);
    float w = dinv[i] * dinv[i];
    agg[idx] = w * hlin[idx];
}

// one wave per edge; lane handles 4 channels (float4 load + 4 f32 atomics)
__global__ void k_scatter128(const long long* __restrict__ ei, const float* __restrict__ hlin,
                             const float* __restrict__ dinv, float* __restrict__ agg, int E) {
    int gtid = blockIdx.x * blockDim.x + threadIdx.x;
    int e = gtid >> 5;
    int lane = threadIdx.x & 31;
    if (e >= E) return;
    long long s = ei[e];
    long long d = ei[(size_t)E + e];
    float w = dinv[s] * dinv[d];
    const float4 v = *(const float4*)(hlin + (size_t)s * HID + lane * 4);
    float* dst = agg + (size_t)d * HID + lane * 4;
    atomicAdd(dst + 0, w * v.x);
    atomicAdd(dst + 1, w * v.y);
    atomicAdd(dst + 2, w * v.z);
    atomicAdd(dst + 3, w * v.w);
}

__global__ void k_init_self8(const float* __restrict__ hh, const float* __restrict__ dinv,
                             float* __restrict__ agg8, int N) {
    size_t idx = (size_t)blockIdx.x * blockDim.x + threadIdx.x;
    if (idx >= (size_t)N * 8) return;
    int i = (int)(idx >> 3);
    float w = dinv[i] * dinv[i];
    agg8[idx] = w * hh[idx];
}

__global__ void k_scatter8(const long long* __restrict__ ei, const float* __restrict__ hh,
                           const float* __restrict__ dinv, float* __restrict__ agg8, int E) {
    int e = blockIdx.x * blockDim.x + threadIdx.x;
    if (e >= E) return;
    long long s = ei[e];
    long long d = ei[(size_t)E + e];
    float w = dinv[s] * dinv[d];
    const float* hs = hh + (size_t)s * 8;
    float* dst = agg8 + (size_t)d * 8;
#pragma unroll
    for (int c = 0; c < 8; ++c) atomicAdd(dst + c, w * hs[c]);
}

// bias + ReLU, emit bf16 directly (input for next WMMA layer)
__global__ void k_bias_relu_bf16(const float* __restrict__ agg, const float* __restrict__ b,
                                 bf16_t* __restrict__ out, int N) {
    size_t idx = (size_t)blockIdx.x * blockDim.x + threadIdx.x;
    if (idx >= (size_t)N * HID) return;
    int c = (int)(idx & 127);
    float v = agg[idx] + b[c];
    out[idx] = (bf16_t)fmaxf(v, 0.0f);
}

// split [N,8] agg into (age[N,3], sex[N,2], eth[N,3]) + biases, flat concat
__global__ void k_writeout(const float* __restrict__ agg8, const float* __restrict__ ba,
                           const float* __restrict__ bs, const float* __restrict__ be,
                           float* __restrict__ out, int N) {
    int i = blockIdx.x * blockDim.x + threadIdx.x;
    if (i >= N) return;
    const float* a = agg8 + (size_t)i * 8;
    float* age = out;
    float* sex = out + (size_t)N * 3;
    float* eth = out + (size_t)N * 5;
    age[i * 3 + 0] = a[0] + ba[0];
    age[i * 3 + 1] = a[1] + ba[1];
    age[i * 3 + 2] = a[2] + ba[2];
    sex[i * 2 + 0] = a[3] + bs[0];
    sex[i * 2 + 1] = a[4] + bs[1];
    eth[i * 3 + 0] = a[5] + be[0];
    eth[i * 3 + 1] = a[6] + be[1];
    eth[i * 3 + 2] = a[7] + be[2];
}

// ---------------------------------------------------------------------------
// Launch
// ---------------------------------------------------------------------------
extern "C" void kernel_launch(void* const* d_in, const int* in_sizes, int n_in,
                              void* d_out, int out_size, void* d_ws, size_t ws_size,
                              hipStream_t stream) {
    const float*     x   = (const float*)d_in[0];
    const long long* ei  = (const long long*)d_in[1];   // int64 edge_index [2,E]
    const float*     W1  = (const float*)d_in[2];
    const float*     b1  = (const float*)d_in[3];
    const float*     W2  = (const float*)d_in[4];
    const float*     b2  = (const float*)d_in[5];
    const float*     Wa  = (const float*)d_in[6];
    const float*     ba  = (const float*)d_in[7];
    const float*     Ws  = (const float*)d_in[8];
    const float*     bs  = (const float*)d_in[9];
    const float*     We  = (const float*)d_in[10];
    const float*     be  = (const float*)d_in[11];

    const int N = in_sizes[0] / HID;      // 50000
    const int E = in_sizes[1] / 2;        // 1,600,000

    // ---- workspace carve-out (all regions fully written before read) ----
    char*  base = (char*)d_ws;
    size_t off  = 0;
    auto bump = [&](size_t bytes) -> char* {
        char* p = base + off;
        off = (off + bytes + 255) & ~(size_t)255;
        return p;
    };
    float*  dinv = (float*)bump((size_t)N * sizeof(float));
    bf16_t* xb   = (bf16_t*)bump((size_t)N * HID * sizeof(bf16_t)); // also reused as h1b/h2b
    float*  hlin = (float*)bump((size_t)N * HID * sizeof(float));   // reused for hh[N,8]
    float*  agg  = (float*)bump((size_t)N * HID * sizeof(float));   // reused for agg8[N,8]
    bf16_t* w1t  = (bf16_t*)bump((size_t)HID * HID * sizeof(bf16_t));
    bf16_t* w2t  = (bf16_t*)bump((size_t)HID * HID * sizeof(bf16_t));
    bf16_t* wht  = (bf16_t*)bump((size_t)16 * HID * sizeof(bf16_t));
    (void)ws_size; (void)n_in; (void)out_size;

    const int T = 256;
    const unsigned gN      = (unsigned)((N + T - 1) / T);
    const unsigned gE      = (unsigned)((E + T - 1) / T);
    const unsigned gNH     = (unsigned)(((size_t)N * HID + T - 1) / T);
    const unsigned gN8     = (unsigned)(((size_t)N * 8 + T - 1) / T);
    const unsigned gEdgeW  = (unsigned)(((size_t)E * 32 + T - 1) / T);   // wave/edge
    const unsigned gTiles  = (unsigned)((N + 15) / 16);                   // 3125
    const unsigned gHead   = (unsigned)((gTiles + 7) / 8);

    // 1) graph prep: dinv
    k_deg_init   <<<gN, T, 0, stream>>>(dinv, N);
    k_deg_acc    <<<gE, T, 0, stream>>>(ei, dinv, E);
    k_deg_to_dinv<<<gN, T, 0, stream>>>(dinv, N);

    // 2) precision prep: bf16 operands (W transposed for contiguous-K B loads)
    k_f32_to_bf16<<<gNH, T, 0, stream>>>(x, xb, (size_t)N * HID);
    k_convW_T    <<<(HID * HID + T - 1) / T, T, 0, stream>>>(W1, w1t);
    k_convW_T    <<<(HID * HID + T - 1) / T, T, 0, stream>>>(W2, w2t);
    k_convHead_T <<<(16 * HID + T - 1) / T, T, 0, stream>>>(Wa, Ws, We, wht);

    // 3) layer 1: WMMA GEMM -> self-loop init -> edge scatter -> bias/ReLU->bf16
    k_gemm128_wmma  <<<gTiles, T, 0, stream>>>(xb, w1t, hlin, N);
    k_init_self128  <<<gNH, T, 0, stream>>>(hlin, dinv, agg, N);
    k_scatter128    <<<gEdgeW, T, 0, stream>>>(ei, hlin, dinv, agg, E);
    k_bias_relu_bf16<<<gNH, T, 0, stream>>>(agg, b1, xb, N);   // xb := h1 (bf16)

    // 4) layer 2
    k_gemm128_wmma  <<<gTiles, T, 0, stream>>>(xb, w2t, hlin, N);
    k_init_self128  <<<gNH, T, 0, stream>>>(hlin, dinv, agg, N);
    k_scatter128    <<<gEdgeW, T, 0, stream>>>(ei, hlin, dinv, agg, E);
    k_bias_relu_bf16<<<gNH, T, 0, stream>>>(agg, b2, xb, N);   // xb := h2 (bf16)

    // 5) fused heads: [N,128]x[128,8] WMMA, then 8-channel conv aggregation
    float* hh   = hlin;   // reuse
    float* agg8 = agg;    // reuse
    k_gemm_head_wmma<<<gHead, T, 0, stream>>>(xb, wht, hh, N);
    k_init_self8    <<<gN8, T, 0, stream>>>(hh, dinv, agg8, N);
    k_scatter8      <<<gE, T, 0, stream>>>(ei, hh, dinv, agg8, E);

    // 6) biases + output layout (age | sex | eth)
    k_writeout<<<gN, T, 0, stream>>>(agg8, ba, bs, be, (float*)d_out, N);
}
